// GeoGAT_56581899157896
// MI455X (gfx1250) — compile-verified, matrix operand order checked by reading, not run on hardware
//
#include <hip/hip_runtime.h>
#include <math.h>

#define NNODES 50000
#define NEDGES 800000
#define EALL   (NEDGES + NNODES)     // edges + self loops
#define NEG_SLOPE_C 0.2f
#define SOFT_EPS 1e-16f
#define BN_EPS_C 1e-5f
#define KDIM 128                     // inner dim of every GEMM
#define LDSTRIDE 136                 // padded LDS row stride (bf16 elems): 272B -> 4-bank lane stride

typedef __attribute__((ext_vector_type(16))) __bf16 v16bf;
typedef __attribute__((ext_vector_type(8)))  __bf16 v8bf;
typedef __attribute__((ext_vector_type(8)))  float  v8f;

// ---------------------------------------------------------------- utilities
__global__ void fill_f32(float* __restrict__ p, float v, long long n) {
    long long t = (long long)blockIdx.x * blockDim.x + threadIdx.x;
    if (t < n) p[t] = v;
}

__global__ void init_bias(float* __restrict__ out, const float* __restrict__ b,
                          long long n, int d) {
    long long t = (long long)blockIdx.x * blockDim.x + threadIdx.x;
    if (t < n) out[t] = b[t % d];
}

__global__ void convert_bf16(const float* __restrict__ in, __bf16* __restrict__ out,
                             long long n) {
    long long t = (long long)blockIdx.x * blockDim.x + threadIdx.x;
    if (t < n) out[t] = (__bf16)in[t];
}

__device__ __forceinline__ float lrelu(float v) {
    return v > 0.f ? v : NEG_SLOPE_C * v;
}

// float atomic max via signed-max/unsigned-min bit trick (valid for mixed signs)
__device__ __forceinline__ void atomicMaxF(float* addr, float value) {
    if (value >= 0.f)
        atomicMax((int*)addr, __float_as_int(value));
    else
        atomicMin((unsigned int*)addr, (unsigned int)__float_as_int(value));
}

// ---------------------------------------------------------------- GEMM (bf16 WMMA, f32 acc)
// C[M x D] = A[M x K=128] * W[128 x D].  Block = 256 threads = 8 waves.
// W staged once per block into LDS, transposed to Wt[n][k] bf16 (<=34KB).
// Each wave owns one 16-row strip and all D/16 col tiles (accumulators in regs).
// Per K-step: A fragment = two global b128 loads (ISA 16-bit A layout gives
// contiguous per-lane k-chunks); ALL B fragments are loaded into a register
// array first so the ds_load_b128s clause up and drain with staggered waits,
// then the 8 WMMAs issue back-to-back.
template <int D>
__global__ void gemm_bf16_wmma(const __bf16* __restrict__ A,
                               const float* __restrict__ W,
                               float* __restrict__ C, int rowTiles) {
    __shared__ __attribute__((aligned(16))) __bf16 wlds[D * LDSTRIDE];

    // stage W: coalesced f32 read W[k][n], bf16 write Wt[n][k]
    for (int i = threadIdx.x; i < KDIM * D; i += 256) {
        int k = i / D, n = i % D;
        wlds[n * LDSTRIDE + k] = (__bf16)W[i];
    }
    __syncthreads();

    const int wave = threadIdx.x >> 5;       // wave-uniform
    const int lane = threadIdx.x & 31;
    const int half = lane >> 4;              // 0: lanes 0-15, 1: lanes 16-31
    const int lidx = lane & 15;
    const int rowTile = blockIdx.x * 8 + wave;
    if (rowTile >= rowTiles) return;         // wave-uniform: EXEC stays all-ones

    constexpr int NCT = D >> 4;              // col tiles per wave
    const __bf16* arow = A + (size_t)(rowTile * 16 + lidx) * KDIM;

    v8f acc[NCT];
#pragma unroll
    for (int ct = 0; ct < NCT; ++ct) acc[ct] = (v8f){};

#pragma unroll
    for (int kt = 0; kt < KDIM; kt += 32) {
        // A fragment: k in [kt+8h, kt+8h+8) and [kt+16+8h, kt+16+8h+8)
        v8bf a0 = *(const v8bf*)(arow + kt + half * 8);
        v8bf a1 = *(const v8bf*)(arow + kt + 16 + half * 8);
        v16bf a;
#pragma unroll
        for (int e = 0; e < 8; ++e) { a[e] = a0[e]; a[8 + e] = a1[e]; }

        // batch all B fragments for this K-step (clause the ds loads)
        v16bf barr[NCT];
#pragma unroll
        for (int ct = 0; ct < NCT; ++ct) {
            const __bf16* wrow = wlds + (ct * 16 + lidx) * LDSTRIDE + kt + half * 8;
            v8bf b0 = *(const v8bf*)(wrow);
            v8bf b1 = *(const v8bf*)(wrow + 16);
#pragma unroll
            for (int e = 0; e < 8; ++e) { barr[ct][e] = b0[e]; barr[ct][8 + e] = b1[e]; }
        }

#pragma unroll
        for (int ct = 0; ct < NCT; ++ct)
            acc[ct] = __builtin_amdgcn_wmma_f32_16x16x32_bf16(
                false, a, false, barr[ct], (short)0, acc[ct], false, false);
    }

#pragma unroll
    for (int ct = 0; ct < NCT; ++ct)
#pragma unroll
        for (int r = 0; r < 8; ++r)
            C[(size_t)(rowTile * 16 + r + half * 8) * D + ct * 16 + lidx] = acc[ct][r];
}

// ---------------------------------------------------------------- attention scores
__global__ void scores_kernel(const float* __restrict__ h,
                              const float* __restrict__ a_s,
                              const float* __restrict__ a_d,
                              float* __restrict__ ss, float* __restrict__ sd,
                              int H, int Cc) {
    long long t = (long long)blockIdx.x * blockDim.x + threadIdx.x;
    if (t >= (long long)NNODES * H) return;
    int n = (int)(t / H), hh = (int)(t % H);
    const float4* hp  = (const float4*)(h   + (size_t)n * H * Cc + (size_t)hh * Cc);
    const float4* asp = (const float4*)(a_s + (size_t)hh * Cc);
    const float4* adp = (const float4*)(a_d + (size_t)hh * Cc);
    float s = 0.f, d = 0.f;
    for (int c = 0; c < (Cc >> 2); ++c) {
        float4 hv = hp[c], av = asp[c], dv = adp[c];
        s += hv.x * av.x + hv.y * av.y + hv.z * av.z + hv.w * av.w;
        d += hv.x * dv.x + hv.y * dv.y + hv.z * dv.z + hv.w * dv.w;
    }
    ss[t] = s; sd[t] = d;
}

// ---------------------------------------------------------------- edge passes
__device__ __forceinline__ void edge_src_dst(const int* ei, int e, int& src, int& dst) {
    if (e < NEDGES) { src = ei[e]; dst = ei[NEDGES + e]; }
    else            { src = e - NEDGES; dst = e - NEDGES; }   // self loops
}

__global__ void edge_max(const int* __restrict__ ei, const float* __restrict__ ss,
                         const float* __restrict__ sd, float* __restrict__ emax, int H) {
    long long t = (long long)blockIdx.x * blockDim.x + threadIdx.x;
    if (t >= (long long)EALL * H) return;
    int e = (int)(t / H), hh = (int)(t % H);
    int src, dst; edge_src_dst(ei, e, src, dst);
    float v = lrelu(ss[(size_t)src * H + hh] + sd[(size_t)dst * H + hh]);
    atomicMaxF(&emax[(size_t)dst * H + hh], v);
}

__global__ void edge_exp(const int* __restrict__ ei, const float* __restrict__ ss,
                         const float* __restrict__ sd, const float* __restrict__ emax,
                         float* __restrict__ p, float* __restrict__ denom, int H) {
    long long t = (long long)blockIdx.x * blockDim.x + threadIdx.x;
    if (t >= (long long)EALL * H) return;
    int e = (int)(t / H), hh = (int)(t % H);
    int src, dst; edge_src_dst(ei, e, src, dst);
    float v = lrelu(ss[(size_t)src * H + hh] + sd[(size_t)dst * H + hh]);
    float pe = expf(v - emax[(size_t)dst * H + hh]);
    p[t] = pe;
    atomicAdd(&denom[(size_t)dst * H + hh], pe);
}

// one thread per (edge, 4 channels): b128 feature load + 4 f32 atomics
__global__ void edge_agg(const int* __restrict__ ei, const float* __restrict__ p,
                         const float* __restrict__ denom, const float* __restrict__ h,
                         float* __restrict__ out, int H, int Cc) {
    int D = H * Cc, Dq = D >> 2;
    long long t = (long long)blockIdx.x * blockDim.x + threadIdx.x;
    if (t >= (long long)EALL * Dq) return;
    int e = (int)(t / Dq), c = ((int)(t % Dq)) << 2, hh = c / Cc;
    int src, dst; edge_src_dst(ei, e, src, dst);
    float alpha = p[(size_t)e * H + hh] / (denom[(size_t)dst * H + hh] + SOFT_EPS);
    float4 hv = *(const float4*)(h + (size_t)src * D + c);
    float* o = out + (size_t)dst * D + c;
    atomicAdd(o + 0, hv.x * alpha);
    atomicAdd(o + 1, hv.y * alpha);
    atomicAdd(o + 2, hv.z * alpha);
    atomicAdd(o + 3, hv.w * alpha);
}

// ---------------------------------------------------------------- batchnorm + ELU (-> bf16 feat)
__global__ void bn_stats(const float* __restrict__ x, float* __restrict__ mean,
                         float* __restrict__ var, int D) {
    __shared__ float s1[256], s2[256];
    int d = blockIdx.x;
    float a = 0.f, b = 0.f;
    for (int n = threadIdx.x; n < NNODES; n += blockDim.x) {
        float v = x[(size_t)n * D + d];
        a += v; b += v * v;
    }
    s1[threadIdx.x] = a; s2[threadIdx.x] = b;
    __syncthreads();
    for (int s = blockDim.x >> 1; s > 0; s >>= 1) {
        if ((int)threadIdx.x < s) {
            s1[threadIdx.x] += s1[threadIdx.x + s];
            s2[threadIdx.x] += s2[threadIdx.x + s];
        }
        __syncthreads();
    }
    if (threadIdx.x == 0) {
        float m = s1[0] / (float)NNODES;
        mean[d] = m;
        var[d]  = s2[0] / (float)NNODES - m * m;
    }
}

__global__ void bn_elu(const float* __restrict__ x, const float* __restrict__ mean,
                       const float* __restrict__ var, const float* __restrict__ g,
                       const float* __restrict__ be, __bf16* __restrict__ out, int D) {
    long long t = (long long)blockIdx.x * blockDim.x + threadIdx.x;
    if (t >= (long long)NNODES * D) return;
    int d = (int)(t % D);
    float v = (x[t] - mean[d]) * rsqrtf(var[d] + BN_EPS_C) * g[d] + be[d];
    out[t] = (__bf16)(v > 0.f ? v : (expf(v) - 1.f));
}

// ---------------------------------------------------------------- host
extern "C" void kernel_launch(void* const* d_in, const int* in_sizes, int n_in,
                              void* d_out, int out_size, void* d_ws, size_t ws_size,
                              hipStream_t stream) {
    (void)in_sizes; (void)n_in; (void)out_size; (void)ws_size;
    const float* x   = (const float*)d_in[0];
    const int*   ei  = (const int*)  d_in[1];
    const float* W0  = (const float*)d_in[2];
    const float* as0 = (const float*)d_in[3];
    const float* ad0 = (const float*)d_in[4];
    const float* g0  = (const float*)d_in[6];
    const float* be0 = (const float*)d_in[7];
    const float* W1  = (const float*)d_in[8];
    const float* as1 = (const float*)d_in[9];
    const float* ad1 = (const float*)d_in[10];
    const float* g1  = (const float*)d_in[12];
    const float* be1 = (const float*)d_in[13];
    const float* W2  = (const float*)d_in[14];
    const float* as2 = (const float*)d_in[15];
    const float* ad2 = (const float*)d_in[16];
    const float* b2  = (const float*)d_in[17];
    float* out = (float*)d_out;

    // workspace layout
    char* ws = (char*)d_ws;
    size_t off = 0;
    auto alloc = [&](size_t bytes) -> void* {
        void* p = (void*)(ws + off);
        off += (bytes + 255) & ~(size_t)255;
        return p;
    };
    float*  B0    = (float*) alloc((size_t)NNODES * 128 * 4);  // h (GEMM out, f32)
    float*  B1    = (float*) alloc((size_t)NNODES * 128 * 4);  // aggregation out (f32)
    __bf16* Fb    = (__bf16*)alloc((size_t)NNODES * 128 * 2);  // bf16 features (GEMM in)
    float*  ss    = (float*) alloc((size_t)NNODES * 4 * 4);
    float*  sd    = (float*) alloc((size_t)NNODES * 4 * 4);
    float*  emax  = (float*) alloc((size_t)NNODES * 4 * 4);
    float*  denom = (float*) alloc((size_t)NNODES * 4 * 4);
    float*  pbuf  = (float*) alloc((size_t)EALL   * 4 * 4);
    float*  mean  = (float*) alloc(128 * 4);
    float*  var   = (float*) alloc(128 * 4);

    auto blocks = [](long long n) -> unsigned int {
        return (unsigned int)((n + 255) / 256);
    };

    const int rowTiles  = NNODES / 16;                 // 3125
    const unsigned gemmBlocks = (unsigned)((rowTiles + 7) / 8);

    // post-GEMM attention + aggregation (shared by all layers)
    auto attn = [&](const float* hbuf, const float* avs, const float* avd,
                    int H, int Cc, float* aggbuf, const float* bias) {
        int D = H * Cc;
        long long nh = (long long)NNODES * H;
        scores_kernel<<<blocks(nh), 256, 0, stream>>>(hbuf, avs, avd, ss, sd, H, Cc);
        fill_f32<<<blocks(nh), 256, 0, stream>>>(emax, -INFINITY, nh);
        fill_f32<<<blocks(nh), 256, 0, stream>>>(denom, 0.f, nh);
        long long neh = (long long)EALL * H;
        edge_max<<<blocks(neh), 256, 0, stream>>>(ei, ss, sd, emax, H);
        edge_exp<<<blocks(neh), 256, 0, stream>>>(ei, ss, sd, emax, pbuf, denom, H);
        long long nd = (long long)NNODES * D;
        if (bias) init_bias<<<blocks(nd), 256, 0, stream>>>(aggbuf, bias, nd, D);
        else      fill_f32 <<<blocks(nd), 256, 0, stream>>>(aggbuf, 0.f, nd);
        edge_agg<<<blocks((long long)EALL * (D >> 2)), 256, 0, stream>>>(
            ei, pbuf, denom, hbuf, aggbuf, H, Cc);
    };

    // ---- layer 0 (bias b0 == 0 and cancels through BN)
    convert_bf16<<<blocks((long long)NNODES * 128), 256, 0, stream>>>(
        x, Fb, (long long)NNODES * 128);
    gemm_bf16_wmma<128><<<gemmBlocks, 256, 0, stream>>>(Fb, W0, B0, rowTiles);
    attn(B0, as0, ad0, 4, 32, B1, nullptr);
    bn_stats<<<128, 256, 0, stream>>>(B1, mean, var, 128);
    bn_elu<<<blocks((long long)NNODES * 128), 256, 0, stream>>>(
        B1, mean, var, g0, be0, Fb, 128);

    // ---- layer 1
    gemm_bf16_wmma<128><<<gemmBlocks, 256, 0, stream>>>(Fb, W1, B0, rowTiles);
    attn(B0, as1, ad1, 4, 32, B1, nullptr);
    bn_stats<<<128, 256, 0, stream>>>(B1, mean, var, 128);
    bn_elu<<<blocks((long long)NNODES * 128), 256, 0, stream>>>(
        B1, mean, var, g1, be1, Fb, 128);

    // ---- layer 2: heads=1, OUT=16, concat=False (mean over 1 head == identity)
    gemm_bf16_wmma<16><<<gemmBlocks, 256, 0, stream>>>(Fb, W2, B0, rowTiles);
    attn(B0, as2, ad2, 1, 16, out, b2);
}
